// RainbowDQNNetwork_20349555048811
// MI455X (gfx1250) — compile-verified, matrix operand order checked by reading, not run on hardware
//
#include <hip/hip_runtime.h>
#include <hip/hip_bf16.h>

typedef __attribute__((ext_vector_type(16))) _Float16 v16h;
typedef __attribute__((ext_vector_type(8)))  _Float16 v8h;
typedef __attribute__((ext_vector_type(8)))  float    v8f;

#define BATCH   524288
#define HID     64
#define N_ATOMS 51
#define N_ACT   2
#define ROWS_PER_BLOCK 128   // 8 waves * 16 rows
#define GRID_MAIN (BATCH / ROWS_PER_BLOCK)   // 4096

// ---------------------------------------------------------------------------
// Kernel 0: build noisy weights packed into WMMA B-operand layout (f16),
// plus noisy biases (f32).
// Packed layout per matrix: [nt(4)][kb(2)][lane(32)][e(16)] halves,
// element = W[K][N], N = nt*16 + lane%16,
// K = kb*32 + (e&7) + 16*(e>>3) + 8*(lane>>4)   (ISA 16-bit A/B layout)
// ---------------------------------------------------------------------------
__global__ __launch_bounds__(256) void rainbow_prep(
    const float* __restrict__ nk, const float* __restrict__ nb,
    const float* __restrict__ K2, const float* __restrict__ S2,
    const float* __restrict__ B2, const float* __restrict__ SB2,
    const float* __restrict__ KV, const float* __restrict__ SV,
    const float* __restrict__ BV, const float* __restrict__ SBV,
    const float* __restrict__ KA, const float* __restrict__ SA,
    const float* __restrict__ BA, const float* __restrict__ SBA,
    _Float16* __restrict__ packedB, float* __restrict__ nbias)
{
    int idx = blockIdx.x * 256 + threadIdx.x;
    if (idx < 3 * 4096) {
        int mat  = idx >> 12;
        int rem  = idx & 4095;
        int pair = rem >> 9;          // nt*2 + kb
        int nt   = pair >> 1;
        int kb   = pair & 1;
        int rem2 = rem & 511;
        int lane = rem2 >> 4;
        int e    = rem2 & 15;
        int N  = nt * 16 + (lane & 15);
        int Kk = kb * 32 + (e & 7) + ((e >> 3) << 4) + ((lane >> 4) << 3);
        const float* Km = (mat == 0) ? K2 : ((mat == 1) ? KV : KA);
        const float* Sm = (mat == 0) ? S2 : ((mat == 1) ? SV : SA);
        int w = Kk * HID + N;
        packedB[idx] = (_Float16)(Km[w] + Sm[w] * nk[w]);
    } else if (idx < 3 * 4096 + 3 * HID) {
        int t   = idx - 3 * 4096;
        int mat = t >> 6;
        int j   = t & 63;
        const float* Bm  = (mat == 0) ? B2  : ((mat == 1) ? BV  : BA);
        const float* SBm = (mat == 0) ? SB2 : ((mat == 1) ? SBV : SBA);
        nbias[t] = Bm[j] + SBm[j] * nb[j];
    }
}

// ---------------------------------------------------------------------------
// Kernel 1: main fused trunk.  256 threads = 8 waves; each wave owns 16 rows.
// ---------------------------------------------------------------------------
__global__ __launch_bounds__(256) void rainbow_main(
    const float* __restrict__ x, const float* __restrict__ W1, const float* __restrict__ b1,
    const _Float16* __restrict__ packedB, const float* __restrict__ nbias,
    const float* __restrict__ WV, const float* __restrict__ bV,
    const float* __restrict__ WA, const float* __restrict__ bA,
    float* __restrict__ a_buf, float* __restrict__ v_buf, float* __restrict__ partials)
{
    __shared__ _Float16 buf0[8 * 16 * HID];   // h2, then reused as hv
    __shared__ _Float16 buf1[8 * 16 * HID];   // ha
    __shared__ float    wsum[8];

    const int lane    = threadIdx.x & 31;
    const int wave    = threadIdx.x >> 5;
    const int rowBase = blockIdx.x * ROWS_PER_BLOCK + wave * 16;
    const int M       = lane & 15;
    const int cbase   = (lane >> 4) << 3;     // 0 or 8 (also m0 of D layout)

    // ---- level 0: h1 = relu(x @ W1 + b1), built directly in A-operand layout
    float xr[4];
#pragma unroll
    for (int j = 0; j < 4; ++j) xr[j] = x[(size_t)(rowBase + M) * 4 + j];

    v16h A1k0, A1k1;
#pragma unroll
    for (int e = 0; e < 16; ++e) {
        int kk = (e & 7) + ((e >> 3) << 4) + cbase;       // K for kb=0; +32 for kb=1
        float a0 = b1[kk], a1 = b1[kk + 32];
#pragma unroll
        for (int j = 0; j < 4; ++j) {
            a0 += xr[j] * W1[j * HID + kk];
            a1 += xr[j] * W1[j * HID + kk + 32];
        }
        A1k0[e] = (_Float16)fmaxf(a0, 0.0f);
        A1k1[e] = (_Float16)fmaxf(a1, 0.0f);
    }

    // ---- level 1: h2 = relu(h1 @ W2n + b2n)  (8 WMMAs)
    _Float16* h2w = buf0 + wave * (16 * HID);
#pragma unroll
    for (int nt = 0; nt < 4; ++nt) {
        float bb = nbias[nt * 16 + M];
        v8f acc;
#pragma unroll
        for (int r = 0; r < 8; ++r) acc[r] = bb;
        v16h Bm0 = *(const v16h*)(packedB + (nt * 2 + 0) * 512 + lane * 16);
        v16h Bm1 = *(const v16h*)(packedB + (nt * 2 + 1) * 512 + lane * 16);
        acc = __builtin_amdgcn_wmma_f32_16x16x32_f16(false, A1k0, false, Bm0, (short)0, acc, false, false);
        acc = __builtin_amdgcn_wmma_f32_16x16x32_f16(false, A1k1, false, Bm1, (short)0, acc, false, false);
#pragma unroll
        for (int r = 0; r < 8; ++r)
            h2w[(cbase + r) * HID + nt * 16 + M] = (_Float16)fmaxf(acc[r], 0.0f);
    }

    // ---- re-layout h2 (D layout in LDS) into A operands (in-order LDS, no barrier)
    union V16U { v16h v; v8h h[2]; };
    v16h A2k0, A2k1;
    {
        V16U t0, t1;
        t0.h[0] = *(const v8h*)(h2w + M * HID + cbase);
        t0.h[1] = *(const v8h*)(h2w + M * HID + cbase + 16);
        t1.h[0] = *(const v8h*)(h2w + M * HID + 32 + cbase);
        t1.h[1] = *(const v8h*)(h2w + M * HID + 32 + cbase + 16);
        A2k0 = t0.v;
        A2k1 = t1.v;
    }

    // ---- level 2: value branch -> buf0 (reuse), advantage branch -> buf1 (16 WMMAs)
    _Float16* haw = buf1 + wave * (16 * HID);
#pragma unroll
    for (int nt = 0; nt < 4; ++nt) {
        float bbv = nbias[64 + nt * 16 + M];
        float bba = nbias[128 + nt * 16 + M];
        v8f accv, acca;
#pragma unroll
        for (int r = 0; r < 8; ++r) { accv[r] = bbv; acca[r] = bba; }
        v16h Bv0 = *(const v16h*)(packedB + 4096 + (nt * 2 + 0) * 512 + lane * 16);
        v16h Bv1 = *(const v16h*)(packedB + 4096 + (nt * 2 + 1) * 512 + lane * 16);
        v16h Ba0 = *(const v16h*)(packedB + 8192 + (nt * 2 + 0) * 512 + lane * 16);
        v16h Ba1 = *(const v16h*)(packedB + 8192 + (nt * 2 + 1) * 512 + lane * 16);
        accv = __builtin_amdgcn_wmma_f32_16x16x32_f16(false, A2k0, false, Bv0, (short)0, accv, false, false);
        accv = __builtin_amdgcn_wmma_f32_16x16x32_f16(false, A2k1, false, Bv1, (short)0, accv, false, false);
        acca = __builtin_amdgcn_wmma_f32_16x16x32_f16(false, A2k0, false, Ba0, (short)0, acca, false, false);
        acca = __builtin_amdgcn_wmma_f32_16x16x32_f16(false, A2k1, false, Ba1, (short)0, acca, false, false);
#pragma unroll
        for (int r = 0; r < 8; ++r) {
            h2w[(cbase + r) * HID + nt * 16 + M] = (_Float16)fmaxf(accv[r], 0.0f);
            haw[(cbase + r) * HID + nt * 16 + M] = (_Float16)fmaxf(acca[r], 0.0f);
        }
    }

    // ---- heads: v = hv @ WV + bV (lanes 0-15), a = ha @ WA + bA (lanes 16-31)
    float s = 0.0f;
    if (lane < 16) {
        float acc = bV[0];
#pragma unroll 8
        for (int n = 0; n < HID; ++n) acc += (float)h2w[lane * HID + n] * WV[n];
        v_buf[rowBase + lane] = acc;
    } else {
        int mm = lane - 16;
        float a0 = bA[0], a1 = bA[1];
#pragma unroll 8
        for (int n = 0; n < HID; ++n) {
            float h = (float)haw[mm * HID + n];
            a0 += h * WA[2 * n];
            a1 += h * WA[2 * n + 1];
        }
        size_t o = (size_t)(rowBase + mm) * 2;
        a_buf[o] = a0;
        a_buf[o + 1] = a1;
        s = a0 + a1;
    }
    // wave-level deterministic sum of a (lanes<16 contribute 0)
#pragma unroll
    for (int off = 16; off >= 1; off >>= 1) s += __shfl_down(s, off, 32);
    if (lane == 0) wsum[wave] = s;
    __syncthreads();
    if (threadIdx.x == 0) {
        float t = 0.0f;
#pragma unroll
        for (int w = 0; w < 8; ++w) t += wsum[w];
        partials[blockIdx.x] = t;
    }
}

// ---------------------------------------------------------------------------
// Kernel 2: deterministic tree reduction of block partials -> sum(a)
// ---------------------------------------------------------------------------
__global__ __launch_bounds__(256) void rainbow_reduce(
    const float* __restrict__ partials, float* __restrict__ sum)
{
    __shared__ float s[256];
    int tid = threadIdx.x;
    float acc = 0.0f;
    for (int i = tid; i < GRID_MAIN; i += 256) acc += partials[i];
    s[tid] = acc;
    __syncthreads();
    for (int off = 128; off >= 1; off >>= 1) {
        if (tid < off) s[tid] += s[tid + off];
        __syncthreads();
    }
    if (tid == 0) sum[0] = s[0];
}

// ---------------------------------------------------------------------------
// Kernel 3: duel + per-action atom heads + softmax; streaming 214 MB write.
// out[a][b][o] with a in {0,1}, o in [0,51)
// ---------------------------------------------------------------------------
__global__ __launch_bounds__(256) void rainbow_out(
    const float* __restrict__ a_buf, const float* __restrict__ v_buf,
    const float* __restrict__ sum,
    const float* __restrict__ Wat, const float* __restrict__ bat,
    float* __restrict__ out)
{
    __shared__ float wat_s[N_ACT * N_ACT * N_ATOMS];  // 204
    __shared__ float bat_s[N_ACT * N_ATOMS];          // 102
    for (int i = threadIdx.x; i < N_ACT * N_ACT * N_ATOMS; i += 256) wat_s[i] = Wat[i];
    for (int i = threadIdx.x; i < N_ACT * N_ATOMS; i += 256) bat_s[i] = bat[i];
    __syncthreads();

    int b = blockIdx.x * 256 + threadIdx.x;
    float mean = sum[0] * (1.0f / (float)(BATCH * N_ACT));
    float vv = v_buf[b];
    float d0 = a_buf[2 * b]     + vv - mean;
    float d1 = a_buf[2 * b + 1] + vv - mean;

#pragma unroll
    for (int act = 0; act < N_ACT; ++act) {
        float l[N_ATOMS];
        float mx = -1e30f;
#pragma unroll
        for (int o = 0; o < N_ATOMS; ++o) {
            float v = d0 * wat_s[act * 102 + o] + d1 * wat_s[act * 102 + 51 + o] + bat_s[act * 51 + o];
            l[o] = v;
            mx = fmaxf(mx, v);
        }
        float ssum = 0.0f;
#pragma unroll
        for (int o = 0; o < N_ATOMS; ++o) { l[o] = __expf(l[o] - mx); ssum += l[o]; }
        float inv = 1.0f / ssum;
        size_t base = (size_t)act * BATCH * N_ATOMS + (size_t)b * N_ATOMS;
#pragma unroll
        for (int o = 0; o < N_ATOMS; ++o) out[base + o] = l[o] * inv;
    }
}

// ---------------------------------------------------------------------------
extern "C" void kernel_launch(void* const* d_in, const int* in_sizes, int n_in,
                              void* d_out, int out_size, void* d_ws, size_t ws_size,
                              hipStream_t stream) {
    const float* x   = (const float*)d_in[0];
    const float* nk  = (const float*)d_in[1];
    const float* nb  = (const float*)d_in[2];
    const float* W1  = (const float*)d_in[3];
    const float* b1  = (const float*)d_in[4];
    const float* K2  = (const float*)d_in[5];
    const float* S2  = (const float*)d_in[6];
    const float* B2  = (const float*)d_in[7];
    const float* SB2 = (const float*)d_in[8];
    const float* KV  = (const float*)d_in[9];
    const float* SV  = (const float*)d_in[10];
    const float* BV  = (const float*)d_in[11];
    const float* SBV = (const float*)d_in[12];
    const float* WV  = (const float*)d_in[13];
    const float* bV  = (const float*)d_in[14];
    const float* KA  = (const float*)d_in[15];
    const float* SA  = (const float*)d_in[16];
    const float* BA  = (const float*)d_in[17];
    const float* SBA = (const float*)d_in[18];
    const float* WA  = (const float*)d_in[19];
    const float* bA  = (const float*)d_in[20];
    const float* Wat = (const float*)d_in[21];
    const float* bat = (const float*)d_in[22];
    float* out = (float*)d_out;

    // workspace carve-up
    char* ws = (char*)d_ws;
    _Float16* packedB = (_Float16*)(ws + 0);                 // 3*4096 halves = 24 KB
    float* nbias      = (float*)(ws + 32768);                // 192 floats
    float* a_buf      = (float*)(ws + 65536);                // BATCH*2 f32 = 4 MB
    float* v_buf      = a_buf + (size_t)BATCH * 2;           // BATCH f32 = 2 MB
    float* partials   = v_buf + BATCH;                       // 4096 f32
    float* sum        = partials + GRID_MAIN;                // 1 f32

    rainbow_prep<<<(3 * 4096 + 3 * HID + 255) / 256, 256, 0, stream>>>(
        nk, nb, K2, S2, B2, SB2, KV, SV, BV, SBV, KA, SA, BA, SBA, packedB, nbias);

    rainbow_main<<<GRID_MAIN, 256, 0, stream>>>(
        x, W1, b1, packedB, nbias, WV, bV, WA, bA, a_buf, v_buf, partials);

    rainbow_reduce<<<1, 256, 0, stream>>>(partials, sum);

    rainbow_out<<<BATCH / 256, 256, 0, stream>>>(a_buf, v_buf, sum, Wat, bat, out);
}
